// SpikingSelfAttention_52312701665561
// MI455X (gfx1250) — 16-bit and 8-bit matrix operand order repaired and checked by reading, compiled, not run on hardware
//
#include <hip/hip_runtime.h>
#include <hip/hip_bf16.h>

typedef __attribute__((ext_vector_type(2)))  unsigned int u32x2;
typedef __attribute__((ext_vector_type(4)))  unsigned int u32x4;
typedef __attribute__((ext_vector_type(8)))  int          i32x8;
typedef __attribute__((ext_vector_type(8)))  float        f32x8;
typedef __attribute__((ext_vector_type(16))) __bf16       bf16x16;

union FragBF16 { u32x4 q[2]; bf16x16 v; };
union FragU8   { u32x4 q[2]; u32x2 d[4]; i32x8 v; };

#define TT 4
#define BB 16
#define NN 512
#define CC 512
#define HH 8
#define DD 64
#define BNROWS 8192
#define NELEM  16777216

__device__ __forceinline__ unsigned short f32_to_bf16_rne(float x) {
    unsigned int u = __float_as_uint(x);
    u += 0x7FFFu + ((u >> 16) & 1u);
    return (unsigned short)(u >> 16);
}

__device__ __forceinline__ void async_g2l_b128(unsigned lds_off, const void* gptr) {
    asm volatile("global_load_async_to_lds_b128 %0, %1, off"
                 :: "v"(lds_off), "v"((unsigned long long)gptr) : "memory");
}
__device__ __forceinline__ void wait_asynccnt0() {
    asm volatile("s_wait_asynccnt 0x0" ::: "memory");
}
__device__ __forceinline__ unsigned lds_addr32(const void* p) {
    return (unsigned)(unsigned long long)p;
}

__global__ void cvt_f32_bf16(const float* __restrict__ in,
                             unsigned short* __restrict__ out, int n) {
    int i = blockIdx.x * blockDim.x + threadIdx.x;
    int stride = gridDim.x * blockDim.x;
    for (; i < n; i += stride) out[i] = f32_to_bf16_rne(in[i]);
}

__global__ __launch_bounds__(256)
void gemm_bn_lif(const unsigned short* __restrict__ X,
                 const unsigned short* __restrict__ Wt,
                 const float* __restrict__ gamma, const float* __restrict__ beta,
                 const float* __restrict__ mean,  const float* __restrict__ var,
                 const float* __restrict__ bias,
                 unsigned char* __restrict__ spk,
                 float* __restrict__ fout,
                 const float* __restrict__ resid,
                 int mode)
{
    const int tid  = threadIdx.x;
    const int lane = tid & 31, wave = tid >> 5;
    const int hi   = lane >> 4, lm = lane & 15;
    const int rowBase = blockIdx.x * 64 + (wave >> 1) * 16;
    const int colBase = blockIdx.y * 64 + (wave & 1) * 32;

    f32x8 acc[TT][2] = {};
    const int r = rowBase + lm;

    for (int kb = 0; kb < CC; kb += 32) {
        FragBF16 a[TT], b[2];
#pragma unroll
        for (int t = 0; t < TT; ++t) {
            const unsigned short* p =
                X + ((size_t)(t * BNROWS + r) * CC + kb + 8 * hi);
            a[t].q[0] = *(const u32x4*)(p);
            a[t].q[1] = *(const u32x4*)(p + 16);
            __builtin_prefetch((const void*)(p + 32), 0, 3);
        }
#pragma unroll
        for (int j = 0; j < 2; ++j) {
            const int n = colBase + 16 * j + lm;
            const unsigned short* p = Wt + ((size_t)n * CC + kb + 8 * hi);
            b[j].q[0] = *(const u32x4*)(p);
            b[j].q[1] = *(const u32x4*)(p + 16);
        }
#pragma unroll
        for (int t = 0; t < TT; ++t)
#pragma unroll
            for (int j = 0; j < 2; ++j)
                acc[t][j] = __builtin_amdgcn_wmma_f32_16x16x32_bf16(
                    false, a[t].v, false, b[j].v, (short)0, acc[t][j],
                    false, false);
    }

#pragma unroll
    for (int j = 0; j < 2; ++j) {
        const int n = colBase + 16 * j + lm;
        float sc = gamma[n] * rsqrtf(var[n] + 1e-5f);
        float sh = beta[n] - mean[n] * sc;
        if (bias) sh += bias[n] * sc;
#pragma unroll
        for (int g = 0; g < 8; ++g) {
            const int rr = rowBase + g + 8 * hi;
            float vmem = 0.f;
#pragma unroll
            for (int t = 0; t < TT; ++t) {
                float y = acc[t][j][g] * sc + sh;
                float hpot = 0.5f * (vmem + y);
                float s = (hpot >= 1.0f) ? 1.0f : 0.0f;
                vmem = hpot * (1.0f - s);
                if (mode == 0) {
                    spk[(size_t)(t * BNROWS + rr) * CC + n] = (unsigned char)s;
                } else if (mode == 2) {
                    size_t idx = (((size_t)t * BB + (rr >> 9)) * CC + n) * NN + (rr & 511);
                    spk[idx] = (unsigned char)s;
                } else {
                    size_t idx = (size_t)(t * BNROWS + rr) * CC + n;
                    fout[idx] = s + resid[idx];
                }
            }
        }
    }
}

__global__ __launch_bounds__(256)
void spiking_attention(const unsigned char* __restrict__ qs,
                       const unsigned char* __restrict__ ks,
                       const unsigned char* __restrict__ vsT,
                       unsigned short* __restrict__ outbf)
{
    __shared__ unsigned char k_lds[TT * 64 * 64];
    __shared__ unsigned char v_lds[TT * 64 * 64];
    __shared__ unsigned char s_lds[8 * 16 * 64];

    const int tid  = threadIdx.x;
    const int lane = tid & 31, wave = tid >> 5;
    const int hi   = lane >> 4, lm = lane & 15;
    const int b = blockIdx.x >> 3, h = blockIdx.x & 7;
    const int nbase = blockIdx.y * 128 + wave * 16;
    const size_t baseBH = (size_t)b * NN * CC + h * DD;

    FragU8 qf[TT];
    {
        const int n = nbase + lm;
#pragma unroll
        for (int t = 0; t < TT; ++t) {
            const unsigned char* p =
                qs + (size_t)t * BNROWS * CC + baseBH + (size_t)n * CC;
            qf[t].q[0] = *(const u32x4*)(p + 16 * hi);
            qf[t].q[1] = *(const u32x4*)(p + 32 + 16 * hi);
        }
    }

    i32x8 o[TT][4] = {};

    const int st = tid >> 6, srow = tid & 63;
    const unsigned char* kbase_g =
        ks + (size_t)st * BNROWS * CC + baseBH + (size_t)srow * CC;
    const unsigned char* vbase_g =
        vsT + (((size_t)st * BB + b) * CC + h * DD + srow) * NN;
    const unsigned kdst = lds_addr32(&k_lds[(st * 64 + srow) * 64]);
    const unsigned vdst = lds_addr32(&v_lds[(st * 64 + srow) * 64]);

    for (int mc = 0; mc < NN / 64; ++mc) {
        const int mbase = mc * 64;
        {
            const unsigned char* pk = kbase_g + (size_t)mbase * CC;
            const unsigned char* pv = vbase_g + mbase;
#pragma unroll
            for (int i = 0; i < 4; ++i)
                async_g2l_b128(kdst + 16 * i, pk + 16 * i);
#pragma unroll
            for (int i = 0; i < 4; ++i)
                async_g2l_b128(vdst + 16 * i, pv + 16 * i);
            wait_asynccnt0();
        }
        __syncthreads();

        float vmem[4][8];
#pragma unroll
        for (int jm = 0; jm < 4; ++jm)
#pragma unroll
            for (int g = 0; g < 8; ++g) vmem[jm][g] = 0.f;

#pragma unroll
        for (int t = 0; t < TT; ++t) {
            i32x8 aacc[4];
#pragma unroll
            for (int jm = 0; jm < 4; ++jm) {
                FragU8 kf;
                const unsigned char* p = &k_lds[(t * 64 + 16 * jm + lm) * 64 + 16 * hi];
                kf.q[0] = *(const u32x4*)(p);
                kf.q[1] = *(const u32x4*)(p + 32);
                i32x8 z = {};
                aacc[jm] = __builtin_amdgcn_wmma_i32_16x16x64_iu8(
                    false, qf[t].v, false, kf.v, z, false, false);
            }
#pragma unroll
            for (int jm = 0; jm < 4; ++jm)
#pragma unroll
                for (int g = 0; g < 8; ++g) {
                    float y = 0.125f * (float)aacc[jm][g];
                    float hpot = 0.5f * (vmem[jm][g] + y);
                    float s = (hpot >= 0.5f) ? 1.0f : 0.0f;
                    vmem[jm][g] = hpot * (1.0f - s);
                    s_lds[wave * 1024 + (g + 8 * hi) * 64 + 16 * jm + lm] =
                        (unsigned char)s;
                }
            FragU8 af;
            const unsigned char* sp = &s_lds[wave * 1024 + lm * 64];
            af.q[0] = *(const u32x4*)(sp + 16 * hi);
            af.q[1] = *(const u32x4*)(sp + 32 + 16 * hi);
#pragma unroll
            for (int jd = 0; jd < 4; ++jd) {
                FragU8 vf;
                const unsigned char* p = &v_lds[(t * 64 + 16 * jd + lm) * 64 + 16 * hi];
                vf.q[0] = *(const u32x4*)(p);
                vf.q[1] = *(const u32x4*)(p + 32);
                o[t][jd] = __builtin_amdgcn_wmma_i32_16x16x64_iu8(
                    false, af.v, false, vf.v, o[t][jd], false, false);
            }
        }
        __syncthreads();
    }

#pragma unroll
    for (int t = 0; t < TT; ++t)
#pragma unroll
        for (int jd = 0; jd < 4; ++jd)
#pragma unroll
            for (int g = 0; g < 8; ++g) {
                const int n = nbase + g + 8 * hi;
                const int c = h * DD + 16 * jd + lm;
                size_t idx = (size_t)(t * BNROWS + b * NN + n) * CC + c;
                outbf[idx] = f32_to_bf16_rne((float)o[t][jd][g]);
            }
}

extern "C" void kernel_launch(void* const* d_in, const int* in_sizes, int n_in,
                              void* d_out, int out_size, void* d_ws, size_t ws_size,
                              hipStream_t stream) {
    (void)in_sizes; (void)n_in; (void)out_size; (void)ws_size;
    const float* x  = (const float*)d_in[0];
    const float* qw = (const float*)d_in[1];
    const float* qg = (const float*)d_in[2];  const float* qb = (const float*)d_in[3];
    const float* qm = (const float*)d_in[4];  const float* qv = (const float*)d_in[5];
    const float* kw = (const float*)d_in[6];
    const float* kg = (const float*)d_in[7];  const float* kb = (const float*)d_in[8];
    const float* km = (const float*)d_in[9];  const float* kv = (const float*)d_in[10];
    const float* vw = (const float*)d_in[11];
    const float* vg = (const float*)d_in[12]; const float* vb = (const float*)d_in[13];
    const float* vm = (const float*)d_in[14]; const float* vv = (const float*)d_in[15];
    const float* pw = (const float*)d_in[16];
    const float* pg = (const float*)d_in[17]; const float* pb = (const float*)d_in[18];
    const float* pm = (const float*)d_in[19]; const float* pv = (const float*)d_in[20];
    const float* pbias = (const float*)d_in[21];
    float* out = (float*)d_out;

    char* ws = (char*)d_ws;
    const size_t W_ELEMS = (size_t)CC * CC;
    unsigned short* x_bf  = (unsigned short*)(ws);
    unsigned short* qw_bf = (unsigned short*)(ws + 33554432);
    unsigned short* kw_bf = (unsigned short*)(ws + 33554432 + 1 * 524288);
    unsigned short* vw_bf = (unsigned short*)(ws + 33554432 + 2 * 524288);
    unsigned short* pw_bf = (unsigned short*)(ws + 33554432 + 3 * 524288);
    unsigned char*  q_spk = (unsigned char*)(ws + 35651584);
    unsigned char*  k_spk = (unsigned char*)(ws + 52428800);
    unsigned char*  v_spkT= (unsigned char*)(ws + 69206016);
    unsigned short* a_out = (unsigned short*)(ws + 85983232);

    cvt_f32_bf16<<<4096, 256, 0, stream>>>(x,  x_bf,  NELEM);
    cvt_f32_bf16<<<512,  256, 0, stream>>>(qw, qw_bf, (int)W_ELEMS);
    cvt_f32_bf16<<<512,  256, 0, stream>>>(kw, kw_bf, (int)W_ELEMS);
    cvt_f32_bf16<<<512,  256, 0, stream>>>(vw, vw_bf, (int)W_ELEMS);
    cvt_f32_bf16<<<512,  256, 0, stream>>>(pw, pw_bf, (int)W_ELEMS);

    dim3 g1(BNROWS / 64, CC / 64);
    gemm_bn_lif<<<g1, 256, 0, stream>>>(x_bf, qw_bf, qg, qb, qm, qv, nullptr,
                                        q_spk, nullptr, nullptr, 0);
    gemm_bn_lif<<<g1, 256, 0, stream>>>(x_bf, kw_bf, kg, kb, km, kv, nullptr,
                                        k_spk, nullptr, nullptr, 0);
    gemm_bn_lif<<<g1, 256, 0, stream>>>(x_bf, vw_bf, vg, vb, vm, vv, nullptr,
                                        v_spkT, nullptr, nullptr, 2);

    dim3 g2(BB * HH, NN / 128);
    spiking_attention<<<g2, 256, 0, stream>>>(q_spk, k_spk, v_spkT, a_out);

    gemm_bn_lif<<<g1, 256, 0, stream>>>(a_out, pw_bf, pg, pb, pm, pv, pbias,
                                        nullptr, out, x, 1);
}
